// ClassifierGNN_22771916604199
// MI455X (gfx1250) — matrix operand order repaired and checked by reading, not compiled, not run on hardware
//
#include <hip/hip_runtime.h>
#include <hip/hip_bf16.h>

typedef __attribute__((ext_vector_type(16))) _Float16 v16h;
typedef __attribute__((ext_vector_type(8)))  float    v8f;

#define NN 768
#define DD 128
#define C0c 256
#define C1c 128
#define BN_EPS 1e-5f
#define NEG 0.01f
#define PAIRCNT 589824.0f

// workspace layout (bytes)
#define WS_W0A   0u          // 64 frags * 512 halves  (A-frags of W0, G0 in MODE1/2)
#define WS_W0B   65536u      // 64 frags * 512 halves  (B-frags of W0^T, MODE0)
#define WS_W1B   131072u     // 64 frags * 512 halves  (B-frags of W1^T, G1)
#define WS_S0SUM 196608u     // 256 f32
#define WS_S0SS  197632u     // 256 f32
#define WS_S1SUM 198656u     // 128 f32
#define WS_S1SS  199168u     // 128 f32
#define WS_A0    199680u     // 256 f32
#define WS_B0    200704u     // 256 f32
#define WS_A1    201728u     // 128 f32
#define WS_B1    202240u     // 128 f32
#define WS_COL   202752u     // 768 f32
#define WS_AGGR  205824u     // 768*128 f32
#define WS_Y0    599040u     // 130*768 f32
#define WS_Y1    998400u     // 65*768 f32

// ---------------- prep: fp32 -> f16 fragment-swizzled weights, zero stats ----------------
__global__ void prep_k(const float* __restrict__ w0, const float* __restrict__ w1,
                       _Float16* __restrict__ W0A, _Float16* __restrict__ W0B,
                       _Float16* __restrict__ W1B, float* __restrict__ stats) {
    const int id = blockIdx.x * 256 + threadIdx.x;    // 384*256 = 98304 = 3*32768
    const int a    = id >> 15;
    const int r    = id & 32767;
    const int frag = r >> 9;
    const int lane = (r >> 4) & 31;
    const int t    = r & 15;
    const int ln = lane & 15, h = lane >> 4;
    if (a == 0) {
        const int mt = frag >> 2, ks = frag & 3;
        const int row = mt * 16 + ln;
        const int k = ks * 32 + h * 16 + t;
        W0A[r] = (_Float16)w0[row * DD + k];
    } else if (a == 1) {
        const int nt = frag >> 2, ks = frag & 3;
        const int row = nt * 16 + ln;          // out channel (column of B)
        const int k = ks * 32 + h * 16 + t;
        W0B[r] = (_Float16)w0[row * DD + k];
    } else {
        const int nt2 = frag >> 3, kc = frag & 7;
        const int row = nt2 * 16 + ln;
        const int k = kc * 32 + h * 8 + ((t < 8) ? t : t + 8);
        W1B[r] = (_Float16)w1[row * C0c + k];
    }
    if (id < 768) stats[id] = 0.0f;  // s0sum|s0ss|s1sum|s1ss contiguous
}

// ---------------- bn finalize: stats -> (scale a, shift b) ----------------
__global__ void finalize_bn_k(const float* __restrict__ sum, const float* __restrict__ ss,
                              const float* __restrict__ g, const float* __restrict__ bt,
                              float* __restrict__ a, float* __restrict__ b, int nch) {
    int t = blockIdx.x * blockDim.x + threadIdx.x;
    if (t < nch) {
        float m  = sum[t] / PAIRCNT;
        float v  = ss[t] / PAIRCNT - m * m;
        float sc = g[t] * rsqrtf(v + BN_EPS);
        a[t] = sc;
        b[t] = bt[t] - m * sc;
    }
}

// ---------------- fused edge MLP over all 768x768 pairs ----------------
template <int MODE>
__global__ __launch_bounds__(256) void edge_pass_k(
    const float*    __restrict__ feat,
    const _Float16* __restrict__ W0A,
    const _Float16* __restrict__ W0B,
    const _Float16* __restrict__ W1B,
    const float* __restrict__ a0, const float* __restrict__ b0,
    const float* __restrict__ a1, const float* __restrict__ b1,
    const float* __restrict__ wout, const float* __restrict__ bout,
    float* __restrict__ s0sum, float* __restrict__ s0ss,
    float* __restrict__ s1sum, float* __restrict__ s1ss,
    float* __restrict__ sim)
{
    __shared__ float lsum[C0c];
    __shared__ float lss[C0c];
    if (MODE < 2) {
        for (int t = threadIdx.x; t < C0c; t += 256) { lsum[t] = 0.0f; lss[t] = 0.0f; }
        __syncthreads();
    }

    const int lane = threadIdx.x & 31;
    const int w    = threadIdx.x >> 5;
    const int h    = lane >> 4;
    const int ln   = lane & 15;
    const int i0   = blockIdx.y * 16;
    const int j0   = blockIdx.x * 16;
    const float* fi = feat + (i0 + ln) * DD;   // this lane's i-row (pair = ln)

    float sA[8], sS[8];                         // MODE 1 per-lane channel partials
    if (MODE == 1) {
        #pragma unroll
        for (int q = 0; q < 8; q++) { sA[q] = 0.0f; sS[q] = 0.0f; }
    }

    #pragma unroll 1
    for (int jj = 0; jj < 2; jj++) {
        const int j = j0 + w + jj * 8;
        const float* fj = feat + j * DD;

        if (MODE == 0) {
            // ---- A-fragments of V^T (pair-major) ----
            v16h Vt[4];
            #pragma unroll
            for (int ks = 0; ks < 4; ks++) {
                #pragma unroll
                for (int t = 0; t < 16; t++) {
                    const int k = ks * 32 + h * 16 + t;
                    Vt[ks][t] = (_Float16)fabsf(fi[k] - fj[k]);
                }
            }
            #pragma unroll 2
            for (int nt = 0; nt < 16; nt++) {
                const _Float16* bb = W0B + nt * 2048 + lane * 16;
                v16h B0 = *(const v16h*)(bb);
                v16h B1 = *(const v16h*)(bb + 512);
                v16h B2 = *(const v16h*)(bb + 1024);
                v16h B3 = *(const v16h*)(bb + 1536);
                v8f acc;
                #pragma unroll
                for (int r = 0; r < 8; r++) acc[r] = 0.0f;
                acc = __builtin_amdgcn_wmma_f32_16x16x32_f16(false, Vt[0], false, B0, (short)0, acc, false, false);
                acc = __builtin_amdgcn_wmma_f32_16x16x32_f16(false, Vt[1], false, B1, (short)0, acc, false, false);
                acc = __builtin_amdgcn_wmma_f32_16x16x32_f16(false, Vt[2], false, B2, (short)0, acc, false, false);
                acc = __builtin_amdgcn_wmma_f32_16x16x32_f16(false, Vt[3], false, B3, (short)0, acc, false, false);
                float s = 0.0f, ss = 0.0f;
                #pragma unroll
                for (int r = 0; r < 8; r++) { s += acc[r]; ss += acc[r] * acc[r]; }
                s  += __shfl_xor(s, 16);
                ss += __shfl_xor(ss, 16);
                if (lane < 16) {                          // channel = nt*16 + lane
                    atomicAdd(&lsum[nt * 16 + lane], s);
                    atomicAdd(&lss[nt * 16 + lane], ss);
                }
            }
        } else {
            // ---- B-fragments of V (channel-major rows) ----
            v16h Bf[4];
            #pragma unroll
            for (int ks = 0; ks < 4; ks++) {
                const int cb = ks * 32 + h * 16;
                #pragma unroll
                for (int t = 0; t < 16; t++)
                    Bf[ks][t] = (_Float16)fabsf(fi[cb + t] - fj[cb + t]);
            }

            v8f acc2[8];                                  // G1 accumulators (z1^T tiles)
            #pragma unroll
            for (int q = 0; q < 8; q++) {
                #pragma unroll
                for (int r = 0; r < 8; r++) acc2[q][r] = 0.0f;
            }

            #pragma unroll 1
            for (int mt2 = 0; mt2 < 8; mt2++) {           // K-chunks of G1
                v16h hAc;                                 // h0^T A-fragment (in-lane)
                #pragma unroll
                for (int hf = 0; hf < 2; hf++) {
                    const int mt = mt2 * 2 + hf;
                    const _Float16* ap = W0A + mt * 2048 + lane * 16;
                    v16h A0 = *(const v16h*)(ap);
                    v16h A1 = *(const v16h*)(ap + 512);
                    v16h A2 = *(const v16h*)(ap + 1024);
                    v16h A3 = *(const v16h*)(ap + 1536);
                    v8f acc;
                    #pragma unroll
                    for (int r = 0; r < 8; r++) acc[r] = 0.0f;
                    acc = __builtin_amdgcn_wmma_f32_16x16x32_f16(false, A0, false, Bf[0], (short)0, acc, false, false);
                    acc = __builtin_amdgcn_wmma_f32_16x16x32_f16(false, A1, false, Bf[1], (short)0, acc, false, false);
                    acc = __builtin_amdgcn_wmma_f32_16x16x32_f16(false, A2, false, Bf[2], (short)0, acc, false, false);
                    acc = __builtin_amdgcn_wmma_f32_16x16x32_f16(false, A3, false, Bf[3], (short)0, acc, false, false);
                    // bn0 + lrelu -> f16 halves of hAc
                    const int cb = mt * 16 + h * 8;
                    const float4 A01 = *(const float4*)(a0 + cb);
                    const float4 A23 = *(const float4*)(a0 + cb + 4);
                    const float4 B01 = *(const float4*)(b0 + cb);
                    const float4 B23 = *(const float4*)(b0 + cb + 4);
                    const float av[8] = {A01.x, A01.y, A01.z, A01.w, A23.x, A23.y, A23.z, A23.w};
                    const float bv[8] = {B01.x, B01.y, B01.z, B01.w, B23.x, B23.y, B23.z, B23.w};
                    #pragma unroll
                    for (int r = 0; r < 8; r++) {
                        float y = av[r] * acc[r] + bv[r];
                        y = (y >= 0.0f) ? y : NEG * y;
                        hAc[hf * 8 + r] = (_Float16)y;
                    }
                }
                // G1 partial products for K-chunk kc = mt2
                const _Float16* wb = W1B + mt2 * 512 + lane * 16;
                #pragma unroll
                for (int nt2 = 0; nt2 < 8; nt2++) {
                    v16h Bg = *(const v16h*)(wb + nt2 * 4096);
                    acc2[nt2] = __builtin_amdgcn_wmma_f32_16x16x32_f16(false, hAc, false, Bg, (short)0, acc2[nt2], false, false);
                }
            }

            if (MODE == 1) {
                #pragma unroll
                for (int nt2 = 0; nt2 < 8; nt2++) {
                    float s = 0.0f, ss = 0.0f;
                    #pragma unroll
                    for (int r = 0; r < 8; r++) { s += acc2[nt2][r]; ss += acc2[nt2][r] * acc2[nt2][r]; }
                    sA[nt2] += s; sS[nt2] += ss;
                }
            } else {
                float pAcc[8];
                #pragma unroll
                for (int r = 0; r < 8; r++) pAcc[r] = 0.0f;
                #pragma unroll
                for (int nt2 = 0; nt2 < 8; nt2++) {
                    const int c = nt2 * 16 + ln;
                    const float av = a1[c], bv = b1[c], wv = wout[c];
                    #pragma unroll
                    for (int r = 0; r < 8; r++) {
                        float y = av * acc2[nt2][r] + bv;
                        y = (y >= 0.0f) ? y : NEG * y;
                        pAcc[r] += wv * y;
                    }
                }
                const float bo = bout[0];
                #pragma unroll
                for (int r = 0; r < 8; r++) {
                    float p = pAcc[r];
                    p += __shfl_xor(p, 1);
                    p += __shfl_xor(p, 2);
                    p += __shfl_xor(p, 4);
                    p += __shfl_xor(p, 8);
                    if (ln == 0) {
                        const int il = h * 8 + r;               // pair i_local
                        sim[(i0 + il) * NN + j] = 1.0f / (1.0f + expf(-(p + bo)));
                    }
                }
            }
        }
    }

    if (MODE == 1) {
        #pragma unroll
        for (int nt2 = 0; nt2 < 8; nt2++) {
            float s = sA[nt2], ss = sS[nt2];
            s  += __shfl_xor(s, 16);
            ss += __shfl_xor(ss, 16);
            if (lane < 16) {
                atomicAdd(&lsum[nt2 * 16 + lane], s);
                atomicAdd(&lss[nt2 * 16 + lane], ss);
            }
        }
    }
    if (MODE < 2) {
        __syncthreads();
        const int nch = (MODE == 0) ? C0c : C1c;
        float* gs  = (MODE == 0) ? s0sum : s1sum;
        float* gss = (MODE == 0) ? s0ss  : s1ss;
        for (int t = threadIdx.x; t < nch; t += 256) {
            atomicAdd(&gs[t],  lsum[t]);
            atomicAdd(&gss[t], lss[t]);
        }
    }
}

// ---------------- edge column sums ----------------
__global__ void colsum_k(const float* __restrict__ sim, const int* __restrict__ lab,
                         float* __restrict__ colsum) {
    __shared__ int slab[NN];
    for (int t = threadIdx.x; t < NN; t += 256) slab[t] = lab[t];
    __syncthreads();
    const int j  = blockIdx.x * 256 + threadIdx.x;
    const int lj = slab[j];
    float s = 0.0f;
    for (int i = 0; i < NN; i++) {
        float e = (slab[i] == lj) ? sim[i * NN + j] : 0.0f;
        e += 1e-6f + ((i == j) ? 1.0f : 0.0f);
        s += e;
    }
    colsum[j] = s;
}

// ---------------- row normalize + aggregation: aggr = e @ feat ----------------
__global__ __launch_bounds__(128) void aggr_k(const float* __restrict__ sim,
                                              const int* __restrict__ lab,
                                              const float* __restrict__ colsum,
                                              const float* __restrict__ feat,
                                              float* __restrict__ aggr) {
    __shared__ float er[NN];
    __shared__ float red[4];
    const int i  = blockIdx.x;
    const int li = lab[i];
    float part = 0.0f;
    for (int jx = threadIdx.x; jx < NN; jx += 128) {
        float e = (lab[jx] == li) ? sim[i * NN + jx] : 0.0f;
        e += 1e-6f + ((jx == i) ? 1.0f : 0.0f);
        e /= colsum[jx];
        if (jx == i) e = 0.0f;
        er[jx] = e;
        part += fabsf(e);
    }
    part += __shfl_xor(part, 1);
    part += __shfl_xor(part, 2);
    part += __shfl_xor(part, 4);
    part += __shfl_xor(part, 8);
    part += __shfl_xor(part, 16);
    if ((threadIdx.x & 31) == 0) red[threadIdx.x >> 5] = part;
    __syncthreads();
    float rs = fmaxf(red[0] + red[1] + red[2] + red[3], 1e-12f);
    const int c = threadIdx.x;   // 128 channels
    float acc = 0.0f;
    for (int jx = 0; jx < NN; jx++) acc += er[jx] * feat[jx * DD + c];
    aggr[i * DD + c] = acc / rs;
}

// ---------------- node MLP layer 0 ----------------
__global__ __launch_bounds__(160) void n1_k(const float* __restrict__ feat,
                                            const float* __restrict__ aggr,
                                            const float* __restrict__ w0,
                                            float* __restrict__ y0) {
    __shared__ float x[256];
    const int n = blockIdx.x;
    for (int t = threadIdx.x; t < DD; t += 160) {
        x[t]      = feat[n * DD + t];
        x[DD + t] = aggr[n * DD + t];
    }
    __syncthreads();
    const int o = threadIdx.x;
    if (o < 130) {
        float s = 0.0f;
        for (int c = 0; c < 256; c++) s += w0[o * 256 + c] * x[c];
        y0[o * NN + n] = s;
    }
}

// ---------------- per-row BN, optional lrelu / transposed output ----------------
template <bool LRELU, bool TRANS_OUT>
__global__ void bn_rows_k(float* __restrict__ y, const float* __restrict__ g,
                          const float* __restrict__ bt, float* __restrict__ outT, int ocols) {
    __shared__ float r1[8], r2[8];
    const int o = blockIdx.x;
    float s = 0.0f, ss = 0.0f;
    for (int n = threadIdx.x; n < NN; n += 256) {
        float v = y[o * NN + n];
        s += v; ss += v * v;
    }
    s  += __shfl_xor(s, 1);  ss += __shfl_xor(ss, 1);
    s  += __shfl_xor(s, 2);  ss += __shfl_xor(ss, 2);
    s  += __shfl_xor(s, 4);  ss += __shfl_xor(ss, 4);
    s  += __shfl_xor(s, 8);  ss += __shfl_xor(ss, 8);
    s  += __shfl_xor(s, 16); ss += __shfl_xor(ss, 16);
    if ((threadIdx.x & 31) == 0) { r1[threadIdx.x >> 5] = s; r2[threadIdx.x >> 5] = ss; }
    __syncthreads();
    float S = 0.0f, SS = 0.0f;
    #pragma unroll
    for (int k = 0; k < 8; k++) { S += r1[k]; SS += r2[k]; }
    const float m  = S / (float)NN;
    const float vr = SS / (float)NN - m * m;
    const float a  = g[o] * rsqrtf(vr + BN_EPS);
    const float b  = bt[o] - m * a;
    for (int n = threadIdx.x; n < NN; n += 256) {
        float v = a * y[o * NN + n] + b;
        if (LRELU) v = (v >= 0.0f) ? v : NEG * v;
        if (TRANS_OUT) outT[n * ocols + o] = v;
        else           y[o * NN + n] = v;
    }
}

// ---------------- node MLP layer 1 ----------------
__global__ __launch_bounds__(96) void n3_k(const float* __restrict__ y0,
                                           const float* __restrict__ w1,
                                           float* __restrict__ y1) {
    __shared__ float xc[130];
    const int n = blockIdx.x;
    for (int t = threadIdx.x; t < 130; t += 96) xc[t] = y0[t * NN + n];
    __syncthreads();
    const int o = threadIdx.x;
    if (o < 65) {
        float s = 0.0f;
        for (int c = 0; c < 130; c++) s += w1[o * 130 + c] * xc[c];
        y1[o * NN + n] = s;
    }
}

extern "C" void kernel_launch(void* const* d_in, const int* in_sizes, int n_in,
                              void* d_out, int out_size, void* d_ws, size_t ws_size,
                              hipStream_t stream) {
    const float* feat  = (const float*)d_in[0];
    const int*   lab   = (const int*)  d_in[1];
    const float* e_w0  = (const float*)d_in[2];
    const float* e_g0  = (const float*)d_in[3];
    const float* e_b0  = (const float*)d_in[4];
    const float* e_w1  = (const float*)d_in[5];
    const float* e_g1  = (const float*)d_in[6];
    const float* e_b1  = (const float*)d_in[7];
    const float* e_wout= (const float*)d_in[8];
    const float* e_bout= (const float*)d_in[9];
    const float* n_w0  = (const float*)d_in[10];
    const float* n_g0  = (const float*)d_in[11];
    const float* n_b0  = (const float*)d_in[12];
    const float* n_w1  = (const float*)d_in[13];
    const float* n_g1  = (const float*)d_in[14];
    const float* n_b1  = (const float*)d_in[15];

    char* ws = (char*)d_ws;
    _Float16* W0A = (_Float16*)(ws + WS_W0A);
    _Float16* W0B = (_Float16*)(ws + WS_W0B);
    _Float16* W1B = (_Float16*)(ws + WS_W1B);
    float* s0sum = (float*)(ws + WS_S0SUM);
    float* s0ss  = (float*)(ws + WS_S0SS);
    float* s1sum = (float*)(ws + WS_S1SUM);
    float* s1ss  = (float*)(ws + WS_S1SS);
    float* a0    = (float*)(ws + WS_A0);
    float* b0    = (float*)(ws + WS_B0);
    float* a1    = (float*)(ws + WS_A1);
    float* b1    = (float*)(ws + WS_B1);
    float* colsum= (float*)(ws + WS_COL);
    float* aggr  = (float*)(ws + WS_AGGR);
    float* y0    = (float*)(ws + WS_Y0);
    float* y1    = (float*)(ws + WS_Y1);

    float* logits = (float*)d_out;            // 768*65
    float* sim    = (float*)d_out + 49920;    // 768*768

    dim3 pg(48, 48);

    prep_k<<<384, 256, 0, stream>>>(e_w0, e_w1, W0A, W0B, W1B, s0sum);

    edge_pass_k<0><<<pg, 256, 0, stream>>>(feat, W0A, W0B, W1B, a0, b0, a1, b1, e_wout, e_bout,
                                           s0sum, s0ss, s1sum, s1ss, sim);
    finalize_bn_k<<<1, 256, 0, stream>>>(s0sum, s0ss, e_g0, e_b0, a0, b0, 256);

    edge_pass_k<1><<<pg, 256, 0, stream>>>(feat, W0A, W0B, W1B, a0, b0, a1, b1, e_wout, e_bout,
                                           s0sum, s0ss, s1sum, s1ss, sim);
    finalize_bn_k<<<1, 128, 0, stream>>>(s1sum, s1ss, e_g1, e_b1, a1, b1, 128);

    edge_pass_k<2><<<pg, 256, 0, stream>>>(feat, W0A, W0B, W1B, a0, b0, a1, b1, e_wout, e_bout,
                                           s0sum, s0ss, s1sum, s1ss, sim);

    colsum_k<<<3, 256, 0, stream>>>(sim, lab, colsum);
    aggr_k<<<NN, 128, 0, stream>>>(sim, lab, colsum, feat, aggr);

    n1_k<<<NN, 160, 0, stream>>>(feat, aggr, n_w0, y0);
    bn_rows_k<true, false><<<130, 256, 0, stream>>>(y0, n_g0, n_b0, nullptr, 0);
    n3_k<<<NN, 96, 0, stream>>>(y0, n_w1, y1);
    bn_rows_k<false, true><<<65, 256, 0, stream>>>(y1, n_g1, n_b1, logits, 65);
}
